// DMel_6571299963221
// MI455X (gfx1250) — matrix operand order repaired and checked by reading, not compiled, not run on hardware
//
#include <hip/hip_runtime.h>

typedef __attribute__((ext_vector_type(16))) _Float16 v16h;
typedef __attribute__((ext_vector_type(8)))  _Float16 v8h;
typedef __attribute__((ext_vector_type(8)))  float    v8f;

#define NFFT    1024
#define HOP     320
#define LSAMP   320000
#define NFRAMES 1001
#define NFREQ   513
#define NPAD    576      // freq bins padded to 9 * 64
#define BATCH   32

#define KC 64            // K chunk staged in LDS per iteration
#define KS 72            // LDS k-stride in halves (144B rows, 16B-aligned, conflict-padded)
#define MT 128           // block tile: frames (8 waves x 16)
#define NT 64            // block tile: freq bins (4 WMMA subtiles per wave)

// ---------- CDNA5 TDM availability / arity detection ----------
#ifndef __has_builtin
#define __has_builtin(x) 0
#endif
#if __has_builtin(__builtin_amdgcn_tensor_load_to_lds) && \
    __has_builtin(__builtin_amdgcn_s_wait_tensorcnt)
#define HAVE_TDM 1
#else
#define HAVE_TDM 0
#endif
#if __has_include(<hip/amd_detail/amd_gfx1250_TDM.h>)
#define TDM_SIX_ARG 1    // therock-10.0 headers -> 6-arg builtin
#else
#define TDM_SIX_ARG 0    // ROCm 7.2 -> 5-arg builtin
#endif

#if HAVE_TDM
typedef unsigned int u32x4 __attribute__((ext_vector_type(4)));
typedef int          i32x4 __attribute__((ext_vector_type(4)));
typedef int          i32x8 __attribute__((ext_vector_type(8)));

// Async-copy one basis tile: 64 rows x 64 halves, global row stride = 1024 halves,
// LDS row stride = 72 halves via TDM padding (pad_interval=32 dwords -> every 128B row,
// pad_amount=4 dwords -> +16B), per CDNA5 ISA ch.8 D# layout.
__device__ __forceinline__ void tdm_load_tile(const _Float16* gsrc, _Float16* ldst) {
  const unsigned long long ga = (unsigned long long)gsrc;
  const unsigned int       la = (unsigned int)(unsigned long long)ldst;
  u32x4 g0 = { 1u,                                   // count=1 (valid user descriptor)
               la,                                   // lds_addr
               (unsigned int)ga,                     // global_addr[31:0]
               ((unsigned int)(ga >> 32) & 0x01FFFFFFu) | (2u << 30) };  // [56:32]|type=2
  i32x8 g1 = { (int)0x07110000u,                     // data_size=2B, pad_en, intv=32dw, amt=4dw
               (int)((1024u & 0xFFFFu) << 16),       // tensor_dim0 lo16  -> bits[63:48]
               (int)((1024u >> 16) | ((unsigned)NPAD << 16)),  // dim0 hi | dim1 lo
               (int)(((unsigned)NPAD >> 16) | (64u << 16)),    // dim1 hi | tile_dim0=64
               (int)64,                              // tile_dim1=64 rows, tile_dim2=0
               (int)1024,                            // tensor_dim0_stride lo32 (elements)
               0, 0 };
  i32x4 z4 = { 0, 0, 0, 0 };
#if TDM_SIX_ARG
  i32x8 z8 = { 0, 0, 0, 0, 0, 0, 0, 0 };
  __builtin_amdgcn_tensor_load_to_lds(g0, g1, z4, z4, z8, 0);
#else
  __builtin_amdgcn_tensor_load_to_lds(g0, g1, z4, z4, 0);
#endif
}
#endif  // HAVE_TDM

// ---------------- prep: per-batch mean (DC removal) ----------------
__global__ __launch_bounds__(256) void mean_kernel(const float* __restrict__ x,
                                                   float* __restrict__ mean) {
  const int b = blockIdx.x;
  const float* xb = x + (size_t)b * LSAMP;
  float s = 0.f;
  for (int i = threadIdx.x; i < LSAMP; i += 256) s += xb[i];
  __shared__ float red[256];
  red[threadIdx.x] = s;
  __syncthreads();
  for (int st = 128; st > 0; st >>= 1) {
    if (threadIdx.x < st) red[threadIdx.x] += red[threadIdx.x + st];
    __syncthreads();
  }
  if (threadIdx.x == 0) mean[b] = red[0] * (1.0f / (float)LSAMP);
}

// ---------------- prep: Hann window + transposed f16 DFT basis ----------------
__global__ __launch_bounds__(256) void basis_kernel(_Float16* __restrict__ cosT,
                                                    _Float16* __restrict__ sinT,
                                                    float* __restrict__ win) {
  const int id = blockIdx.x * 256 + threadIdx.x;   // grid covers NPAD*NFFT exactly
  if (id < NFFT) {
    win[id] = 0.5f * (1.0f - cosf(6.28318530717958647692f * (float)id / (float)NFFT));
  }
  const int n = id >> 10;
  const int k = id & (NFFT - 1);
  const int m = (n * k) & (NFFT - 1);   // exact angle: (n*k) mod 1024
  const float th = (float)m * (6.28318530717958647692f / (float)NFFT);
  float s, c;
  __sincosf(th, &s, &c);
  cosT[id] = (_Float16)c;
  sinT[id] = (_Float16)s;
}

// ---------------- fused frame/GEMM/compress kernel ----------------
// Per batch:  C[T,F] (cos & sin planes) = frames[T,1024] x basis[1024,F]
// out[b,f,t] = log1p(lambda * (re^2 + im^2))
__global__ __launch_bounds__(256) void stft_wmma_kernel(
    const float* __restrict__ x, const float* __restrict__ mean,
    const float* __restrict__ win, const _Float16* __restrict__ cosT,
    const _Float16* __restrict__ sinT, const float* __restrict__ lambd,
    float* __restrict__ out) {
  __shared__ __align__(16) _Float16 sA[MT * KS];          // 18.0 KB windowed frames
  __shared__ __align__(16) _Float16 sB[2][2][NT * KS];    // 36.0 KB double-buffered cos/sin

  const int tid  = threadIdx.x;
  const int b    = blockIdx.z;
  const int t0   = blockIdx.x * MT;
  const int n0   = blockIdx.y * NT;
  const int wave = tid >> 5;            // 8 waves stacked along frames
  const int lane = tid & 31;
  const int t0w  = t0 + wave * 16;

  const float mu = mean[b];
  const float* xb = x + (size_t)b * LSAMP;

  v8f accC[4] = {};
  v8f accS[4] = {};

  // 16-bit WMMA fragment addressing used here (contiguous k, same order for A and B):
  // A 16x32: lane holds row M=lane&15; K halves {base..base+15}, base=lane<16?0:16
  // B 32x16: lane holds col N=lane&15; K halves {kb..kb+15}, kb=lane<16?0:16
  const int arow  = lane & 15;
  const int abase = (lane < 16) ? 0 : 16;
  const int bcol  = lane & 15;
  const int bbase = (lane < 16) ? 0 : 16;

  int cur = 0;
#if HAVE_TDM
  if (wave == 0) {  // wave-uniform branch; TDM issue is once per wave, EXEC-ignored
    tdm_load_tile(cosT + (size_t)n0 * NFFT + 0, &sB[0][0][0]);
    tdm_load_tile(sinT + (size_t)n0 * NFFT + 0, &sB[0][1][0]);
  }
#endif

  for (int kc = 0; kc < NFFT; kc += KC) {
    __syncthreads();   // previous chunk's compute done: safe to refill A / issue into alt B

    // ---- stage A: 128 frames x 64 samples, reflect-padded, DC-removed, windowed
#pragma unroll
    for (int e = 0; e < (MT * KC) / 256; ++e) {        // 32 iters
      const int idx = e * 256 + tid;
      const int fr  = idx >> 6;
      const int k   = idx & 63;
      int p = (t0 + fr) * HOP + kc + k - (NFFT / 2);   // center pad
      p = (p < 0) ? -p : p;                            // reflect left
      p = (p >= LSAMP) ? (2 * LSAMP - 2 - p) : p;      // reflect right
      const float v = (xb[p] - mu) * win[kc + k];
      sA[fr * KS + k] = (_Float16)v;
      if (kc + KC < NFFT)
        __builtin_prefetch(xb + ((p + KC < LSAMP) ? p + KC : p), 0, 0);
    }

#if HAVE_TDM
    if (wave == 0) {
      __builtin_amdgcn_s_wait_tensorcnt(0);            // this chunk's B tiles landed
      if (kc + KC < NFFT) {                            // kick off next chunk into alt buffer
        tdm_load_tile(cosT + (size_t)n0 * NFFT + kc + KC, &sB[cur ^ 1][0][0]);
        tdm_load_tile(sinT + (size_t)n0 * NFFT + kc + KC, &sB[cur ^ 1][1][0]);
      }
    }
#else
    // ---- fallback: synchronous B staging, 16B vector copies
#pragma unroll
    for (int e = 0; e < 4; ++e) {                      // 2 mats * 64 rows * 8 quads / 256
      const int id  = e * 256 + tid;
      const int mat = id >> 9;
      const int r   = (id >> 3) & (NT - 1);
      const int cq  = id & 7;
      const _Float16* src =
          (mat ? sinT : cosT) + (size_t)(n0 + r) * NFFT + kc + cq * 8;
      _Float16* dst = &sB[cur][mat][r * KS + cq * 8];
      *(uint4*)dst = *(const uint4*)src;
      if (kc + KC < NFFT) __builtin_prefetch(src + KC, 0, 0);
    }
#endif
    __syncthreads();

#pragma unroll
    for (int ks = 0; ks < KC; ks += 32) {
      const v8h alo = *(const v8h*)&sA[(wave * 16 + arow) * KS + ks + abase];
      const v8h ahi = *(const v8h*)&sA[(wave * 16 + arow) * KS + ks + abase + 8];
      v16h afrag;
#pragma unroll
      for (int i = 0; i < 8; ++i) { afrag[i] = alo[i]; afrag[8 + i] = ahi[i]; }

#pragma unroll
      for (int j = 0; j < 4; ++j) {
        const int col = j * 16 + bcol;

        const v8h clo = *(const v8h*)&sB[cur][0][col * KS + ks + bbase];
        const v8h chi = *(const v8h*)&sB[cur][0][col * KS + ks + bbase + 8];
        v16h cfrag;
#pragma unroll
        for (int i = 0; i < 8; ++i) { cfrag[i] = clo[i]; cfrag[8 + i] = chi[i]; }
        accC[j] = __builtin_amdgcn_wmma_f32_16x16x32_f16(
            false, afrag, false, cfrag, (short)0, accC[j], false, false);

        const v8h slo = *(const v8h*)&sB[cur][1][col * KS + ks + bbase];
        const v8h shi = *(const v8h*)&sB[cur][1][col * KS + ks + bbase + 8];
        v16h sfrag;
#pragma unroll
        for (int i = 0; i < 8; ++i) { sfrag[i] = slo[i]; sfrag[8 + i] = shi[i]; }
        accS[j] = __builtin_amdgcn_wmma_f32_16x16x32_f16(
            false, afrag, false, sfrag, (short)0, accS[j], false, false);
      }
    }
    cur ^= 1;
  }

  // ---- epilogue: power + log1p compression, store [B,F,T]
  const float lam = lambd[0];
#pragma unroll
  for (int j = 0; j < 4; ++j) {
    const int f  = n0 + j * 16 + (lane & 15);
    const int tb = t0w + ((lane >> 4) << 3);   // C/D layout: lanes 16-31 hold M+8
#pragma unroll
    for (int r = 0; r < 8; ++r) {
      const int t = tb + r;
      if (t < NFRAMES && f < NFREQ) {
        const float re = accC[j][r];
        const float im = accS[j][r];
        const float pw = re * re + im * im;
        out[((size_t)b * NFREQ + f) * NFRAMES + t] = log1pf(lam * pw);
      }
    }
  }
}

extern "C" void kernel_launch(void* const* d_in, const int* in_sizes, int n_in,
                              void* d_out, int out_size, void* d_ws, size_t ws_size,
                              hipStream_t stream) {
  const float* x     = (const float*)d_in[0];
  const float* lambd = (const float*)d_in[1];
  float* out = (float*)d_out;

  // workspace (~2.37 MB): win[1024] f32 | mean[32] f32 | cosT | sinT (f16, [576][1024])
  char* ws = (char*)d_ws;
  float*    win  = (float*)ws;
  float*    mean = win + NFFT;
  _Float16* cosT = (_Float16*)(mean + BATCH);          // offset 4224 B, 16B aligned
  _Float16* sinT = cosT + (size_t)NPAD * NFFT;

  basis_kernel<<<dim3((NPAD * NFFT) / 256), dim3(256), 0, stream>>>(cosT, sinT, win);
  mean_kernel<<<dim3(BATCH), dim3(256), 0, stream>>>(x, mean);

  dim3 grid((NFRAMES + MT - 1) / MT,   // 8 frame tiles (covers 1024)
            (NFREQ + NT - 1) / NT,     // 9 freq tiles (basis padded to 576)
            BATCH);
  stft_wmma_kernel<<<grid, dim3(256), 0, stream>>>(x, mean, win, cosT, sinT, lambd, out);
}
